// StandardRVQOfficial_31602369364118
// MI455X (gfx1250) — matrix operand order repaired and checked by reading, not compiled, not run on hardware
//
#include <hip/hip_runtime.h>

// ---------------------------------------------------------------------------
// Residual VQ fused kernel for MI455X (gfx1250, wave32, WMMA).
//   B=16, T=4096, D=768, L=8, K=128, COMMIT_W=0.25
// Score GEMM via V_WMMA_F32_16X16X32_BF16 with bf16 hi/lo error-split
// (3 products ~= fp32 dot accuracy). Residual kept in LDS across layers.
// ---------------------------------------------------------------------------

#define D_DIM   768
#define KCB     128
#define LQ      8
#define NTOK    65536            // B*T
#define TOK_WG  32               // tokens per workgroup (2 waves x 16)
#define RSTRIDE 772              // 768 + 4 pad (bank-conflict-free row reads)
#define SSTRIDE 132              // 128 + 4 pad for score rows

typedef __attribute__((ext_vector_type(16))) __bf16 v16bf;
typedef __attribute__((ext_vector_type(8)))  float  v8f;

// ---------------------------------------------------------------------------
// Prep: split fp32 codebooks into bf16 hi/lo + per-code squared norms.
// One block per (layer, code) row.  grid = L*K = 1024, block = 128.
// ---------------------------------------------------------------------------
__global__ __launch_bounds__(128) void rvq_prep_cb(
    const float* __restrict__ cb, __bf16* __restrict__ hi,
    __bf16* __restrict__ lo, float* __restrict__ c2) {
  const int row = blockIdx.x;                 // l*K + k
  const float* src = cb + (size_t)row * D_DIM;
  __bf16* h = hi + (size_t)row * D_DIM;
  __bf16* g = lo + (size_t)row * D_DIM;
  float acc = 0.f;
  for (int d = threadIdx.x; d < D_DIM; d += 128) {
    float v = src[d];
    __bf16 vh = (__bf16)v;                    // RNE truncation to bf16
    __bf16 vl = (__bf16)(v - (float)vh);      // residual term
    h[d] = vh;
    g[d] = vl;
    acc += v * v;
  }
  __shared__ float red[128];
  red[threadIdx.x] = acc;
  __syncthreads();
  for (int s = 64; s > 0; s >>= 1) {
    if (threadIdx.x < s) red[threadIdx.x] += red[threadIdx.x + s];
    __syncthreads();
  }
  if (threadIdx.x == 0) c2[row] = red[0];
}

// ---------------------------------------------------------------------------
// Main fused RVQ kernel: 2 waves/WG, each wave owns a 16-token tile.
// ---------------------------------------------------------------------------
__global__ __launch_bounds__(64) void rvq_main(
    const float* __restrict__ x, const float* __restrict__ cb,
    const __bf16* __restrict__ cbh, const __bf16* __restrict__ cbl,
    const float* __restrict__ c2, float* __restrict__ outQ,
    float* __restrict__ outIdx, float* __restrict__ lossPartial, int numWG) {
  __shared__ __align__(16) float r_s[2][16][RSTRIDE];   // fp32 residual
  __shared__ __align__(16) float s_s[2][16][SSTRIDE];   // scores (dot - c2/2)
  __shared__ int   i_s[2][16];                          // chosen code per row
  __shared__ float redf[2][32];
  __shared__ int   redi[2][32];

  const int tid  = threadIdx.x;
  const int wave = tid >> 5;
  const int lane = tid & 31;
  const int hlf  = lane >> 4;        // lane half (A/B K-interleave select)
  const int hrow = lane & 15;        // row (A) / column (B,C) within tile
  const size_t tokBase = (size_t)blockIdx.x * TOK_WG;

  // -------- Phase A: stage x tile (32 x 768 fp32) into padded LDS ----------
  for (int i = tid; i < (TOK_WG * D_DIM) / 4; i += 64) {
    int off = i * 4;
    int row = off / D_DIM;
    int col = off - row * D_DIM;
    float4 v = *(const float4*)(x + tokBase * D_DIM + off);
    *(float4*)&r_s[row >> 4][row & 15][col] = v;
  }
  __syncthreads();

  // ------------------------------ Layer loop -------------------------------
  for (int l = 0; l < LQ; ++l) {
    // 8 column tiles of 16 codes each: 16x16 f32 accumulators
    v8f acc[8];
#pragma unroll
    for (int t = 0; t < 8; ++t)
#pragma unroll
      for (int j = 0; j < 8; ++j) acc[t][j] = 0.f;

    const __bf16* cbh_l = cbh + (size_t)l * KCB * D_DIM;
    const __bf16* cbl_l = cbl + (size_t)l * KCB * D_DIM;

    for (int d0 = 0; d0 < D_DIM; d0 += 32) {
      // Build A fragments (16x32 bf16, hi+lo) from fp32 residual in LDS.
      // A layout (contiguous, matches B): lanes 0-15 hold K = 0..15,
      //                      lanes 16-31 hold K = 16..31 (same rows).
      v16bf ah, al;
      const float* rrow = &r_s[wave][hrow][0];
      const int b1 = d0 + 16 * hlf;       // elements 0..7  -> K b1..b1+7
      const int b2 = d0 + 16 * hlf + 8;   // elements 8..15 -> K b2..b2+7
#pragma unroll
      for (int e = 0; e < 8; ++e) {
        float v0 = rrow[b1 + e];
        __bf16 h0 = (__bf16)v0;
        ah[e] = h0;
        al[e] = (__bf16)(v0 - (float)h0);
        float v1 = rrow[b2 + e];
        __bf16 h1 = (__bf16)v1;
        ah[e + 8] = h1;
        al[e + 8] = (__bf16)(v1 - (float)h1);
      }
      // B layout: lane n<16 = column n, K d0..d0+15; lane n+16 = K d0+16..31.
      const int dB = d0 + 16 * hlf;
#pragma unroll
      for (int t = 0; t < 8; ++t) {
        const int code = t * 16 + hrow;
        v16bf bh = *(const v16bf*)(cbh_l + (size_t)code * D_DIM + dB);
        v16bf bl = *(const v16bf*)(cbl_l + (size_t)code * D_DIM + dB);
        // bf16x3 split: rh*ch + rl*ch + rh*cl  (~fp32-accurate dot)
        acc[t] = __builtin_amdgcn_wmma_f32_16x16x32_bf16(
            false, ah, false, bh, (short)0, acc[t], false, false);
        acc[t] = __builtin_amdgcn_wmma_f32_16x16x32_bf16(
            false, al, false, bh, (short)0, acc[t], false, false);
        acc[t] = __builtin_amdgcn_wmma_f32_16x16x32_bf16(
            false, ah, false, bl, (short)0, acc[t], false, false);
      }
    }

    // ------ Scores to LDS: argmin(-2 dot + |c|^2) == argmax(dot - c2/2) ----
    const float* c2l = c2 + l * KCB;
#pragma unroll
    for (int t = 0; t < 8; ++t) {
      const int code = t * 16 + hrow;               // C tile: N = hrow
      const float hc2 = 0.5f * c2l[code];
#pragma unroll
      for (int j = 0; j < 8; ++j) {
        const int m = j + 8 * hlf;                  // C tile: M = j (+8 hi half)
        s_s[wave][m][code] = acc[t][j] - hc2;
      }
    }
    __syncthreads();

    // ------ Per-row argmax over 128 codes (ties -> lowest index) -----------
    float best = -3.402823466e38f;
    int bi = 0;
    {
      const int base = hlf * 64;
      for (int c = 0; c < 64; ++c) {
        float s = s_s[wave][hrow][base + c];
        if (s > best) { best = s; bi = base + c; }
      }
    }
    redf[wave][lane] = best;
    redi[wave][lane] = bi;
    __syncthreads();
    if (lane < 16) {
      float ob = redf[wave][lane + 16];             // higher codes: strict >
      int oi = redi[wave][lane + 16];
      if (ob > best) { best = ob; bi = oi; }
      i_s[wave][hrow] = bi;
      outIdx[(tokBase + (size_t)wave * 16 + hrow) * LQ + l] = (float)bi;
    }
    __syncthreads();

    // ------ fp32 residual update + commitment loss -------------------------
    // r_new = r - q  =>  (q - r)^2 == r_new^2
    const float* cb_l = cb + (size_t)l * KCB * D_DIM;
    float ls = 0.f;
    for (int m = 0; m < 16; ++m) {
      const float* q = cb_l + (size_t)i_s[wave][m] * D_DIM;
      float* rr = &r_s[wave][m][0];
#pragma unroll
      for (int it = 0; it < 6; ++it) {
        const int d = it * 128 + lane * 4;
        float4 qv = *(const float4*)(q + d);
        float4 rv = *(float4*)(rr + d);
        rv.x -= qv.x; rv.y -= qv.y; rv.z -= qv.z; rv.w -= qv.w;
        *(float4*)(rr + d) = rv;
        ls += rv.x * rv.x + rv.y * rv.y + rv.z * rv.z + rv.w * rv.w;
      }
    }
    redf[wave][lane] = ls;
    __syncthreads();
    if (lane == 0) {
      float s = 0.f;
      for (int k = 0; k < 32; ++k) s += redf[wave][k];  // fixed order
      lossPartial[(size_t)l * (size_t)numWG * 2 + (size_t)blockIdx.x * 2 + wave] = s;
    }
    __syncthreads();
  }

  // -------- Phase C: quantized = x - r_final ------------------------------
  for (int i = tid; i < (TOK_WG * D_DIM) / 4; i += 64) {
    int off = i * 4;
    int row = off / D_DIM;
    int col = off - row * D_DIM;
    float4 xv = *(const float4*)(x + tokBase * D_DIM + off);
    float4 rv = *(const float4*)&r_s[row >> 4][row & 15][col];
    float4 o;
    o.x = xv.x - rv.x; o.y = xv.y - rv.y; o.z = xv.z - rv.z; o.w = xv.w - rv.w;
    *(float4*)(outQ + tokBase * D_DIM + off) = o;
  }
}

// ---------------------------------------------------------------------------
// Deterministic loss finalization: fixed-order serial sum per layer.
// ---------------------------------------------------------------------------
__global__ void rvq_loss_final(const float* __restrict__ partial,
                               float* __restrict__ outLoss, int n) {
  int l = threadIdx.x;
  if (l < LQ) {
    const float* p = partial + (size_t)l * n;
    float s = 0.f;
    for (int i = 0; i < n; ++i) s += p[i];
    outLoss[l] = 0.25f * s / (float)(NTOK * (size_t)D_DIM);
  }
}

extern "C" void kernel_launch(void* const* d_in, const int* in_sizes, int n_in,
                              void* d_out, int out_size, void* d_ws, size_t ws_size,
                              hipStream_t stream) {
  (void)in_sizes; (void)n_in; (void)out_size; (void)ws_size;
  const float* x  = (const float*)d_in[0];   // (B,T,D) fp32
  const float* cb = (const float*)d_in[1];   // (L,K,D) fp32

  // Workspace layout (all 32B-aligned offsets):
  //   cb_hi bf16: L*K*D*2 = 1,572,864 B
  //   cb_lo bf16: 1,572,864 B
  //   c2 fp32:    L*K*4   = 4,096 B
  //   loss partials: L * numWG * 2 * 4 = 131,072 B
  char* ws = (char*)d_ws;
  __bf16* cbh = (__bf16*)(ws);
  __bf16* cbl = (__bf16*)(ws + 1572864);
  float*  c2  = (float*)(ws + 3145728);
  float*  part = (float*)(ws + 3149824);

  float* outQ    = (float*)d_out;                     // 50,331,648 floats
  float* outIdx  = outQ + (size_t)NTOK * D_DIM;       //    524,288 values
  float* outLoss = outIdx + (size_t)NTOK * LQ;        //          8 floats

  rvq_prep_cb<<<LQ * KCB, 128, 0, stream>>>(cb, cbh, cbl, c2);

  const int numWG = NTOK / TOK_WG;                    // 2048
  rvq_main<<<numWG, 64, 0, stream>>>(x, cb, cbh, cbl, c2,
                                     outQ, outIdx, part, numWG);

  rvq_loss_final<<<1, 64, 0, stream>>>(part, outLoss, numWG * 2);
}
